// TBM_GRU_63221918597467
// MI455X (gfx1250) — matrix operand order repaired and checked by reading, not compiled, not run on hardware
//
#include <hip/hip_runtime.h>
#include <hip/hip_bf16.h>

#define Bsz   128
#define Tlen  1440
#define Hdim  512
#define G3    (3*Hdim)
#define NWG   16
#define GPW   96          // gate columns owned per WG (3 * 32)
#define HPW   32          // h columns owned per WG
#define NT    6           // GPW/16 N-tiles
#define KT    16          // Hdim/32 K-steps
#define THREADS 256

typedef __bf16 bf16;
typedef bf16  v16bf __attribute__((ext_vector_type(16)));
typedef bf16  v8bf  __attribute__((ext_vector_type(8)));
typedef float v8f   __attribute__((ext_vector_type(8)));
typedef int   v4i   __attribute__((ext_vector_type(4)));

#if __has_builtin(__builtin_amdgcn_global_load_async_to_lds_b128) && \
    __has_builtin(__builtin_amdgcn_global_store_async_from_lds_b128) && \
    __has_builtin(__builtin_amdgcn_s_wait_asynccnt)
#define HAS_ASYNC_LDS 1
#else
#define HAS_ASYNC_LDS 0
#endif

#if HAS_ASYNC_LDS
// builtin params are int4* in AS1 (global, prints as __device__) and AS3 (LDS)
typedef __attribute__((address_space(1))) v4i* gas1_v4i;
typedef __attribute__((address_space(3))) v4i* las3_v4i;

__device__ inline void async_g2l_b128(const void* g, const void* l) {
  // per-lane 16-byte copy global -> LDS, tracked by ASYNCcnt
  __builtin_amdgcn_global_load_async_to_lds_b128(
      (gas1_v4i)(size_t)g, (las3_v4i)(size_t)l, 0, 0);
}
__device__ inline void async_l2g_b128(void* g, const void* l) {
  // per-lane 16-byte copy LDS -> global, tracked by ASYNCcnt
  __builtin_amdgcn_global_store_async_from_lds_b128(
      (gas1_v4i)(size_t)g, (las3_v4i)(size_t)l, 0, 0);
}
#endif

// ---------------- grid-wide barrier (cluster fast path + atomic fallback) ---
__device__ inline void grid_barrier(unsigned* cnt, unsigned* gen, unsigned nwg, int cid) {
  __syncthreads();
  if (threadIdx.x == 0) {
    __threadfence();
    if (cid != 0) {
      // dispatched as a workgroup cluster: hardware split-barrier (-3)
      __builtin_amdgcn_s_cluster_barrier();
    } else {
      unsigned g = __hip_atomic_load(gen, __ATOMIC_ACQUIRE, __HIP_MEMORY_SCOPE_AGENT);
      unsigned a = __hip_atomic_fetch_add(cnt, 1u, __ATOMIC_ACQ_REL, __HIP_MEMORY_SCOPE_AGENT);
      if (a + 1u == nwg) {
        __hip_atomic_store(cnt, 0u, __ATOMIC_RELAXED, __HIP_MEMORY_SCOPE_AGENT);
        __hip_atomic_fetch_add(gen, 1u, __ATOMIC_RELEASE, __HIP_MEMORY_SCOPE_AGENT);
      } else {
        while (__hip_atomic_load(gen, __ATOMIC_ACQUIRE, __HIP_MEMORY_SCOPE_AGENT) == g)
          __builtin_amdgcn_s_sleep(2);
      }
    }
  }
  __syncthreads();
}

// ---------------- stage 1: tau / trigger scalars + sync reset ---------------
__global__ void prep_kernel(const float* __restrict__ w_ih, float* scal, unsigned* sync) {
  __shared__ float red[THREADS];
  float s = 0.f;
  for (int i = threadIdx.x; i < G3 * 2; i += THREADS) s += w_ih[i];
  red[threadIdx.x] = s;
  __syncthreads();
  for (int off = THREADS / 2; off > 0; off >>= 1) {
    if (threadIdx.x < off) red[threadIdx.x] += red[threadIdx.x + off];
    __syncthreads();
  }
  if (threadIdx.x == 0) {
    float grads   = red[0] / (float)(G3 * 2) - 1.0f;   // mean(w_ih - 1)
    float tau_new = 6.0f + grads * 0.1f;
    int   trig    = (tau_new >= 1.0f) && (tau_new != 6.0f);
    scal[0] = rintf(tau_new * 0.5f);                    // half_tau (round-half-even)
    ((int*)scal)[1] = trig;
    sync[0] = 0u; sync[1] = 0u;                          // barrier state (every call)
  }
}

// ---------------- stage 2: NaN fix + belief_fill (one thread per row) -------
__global__ void fill_kernel(const float* __restrict__ inp, const float* __restrict__ scal,
                            float* __restrict__ xs, int* __restrict__ pb) {
  int b = threadIdx.x;
  if (b >= Bsz) return;
  const float ht   = scal[0];
  const int   trig = ((const int*)scal)[1];
  const float* xrow = inp + (size_t)b * 3 * Tlen;       // channel 0
  const float* mrow = xrow + 2 * Tlen;                  // channel 2 (mask)
  float* xo = xs + (size_t)b * Tlen;
  int*   po = pb + (size_t)b * Tlen;
  int p = -1;
  for (int t = 0; t < Tlen; ++t) {
    float x = xrow[t];
    if (x != x) x = -1.0f;                              // isnan -> -1
    xo[t] = x;
    if (mrow[t] == 1.0f) p = t;
    po[t] = p;
  }
  int n = Tlen;
  for (int t = Tlen - 1; t >= 0; --t) {
    if (mrow[t] == 1.0f) n = t;
    if (trig) {
      float xv = xo[t];
      bool use_n = (n < Tlen) && ((float)t >= (float)n - ht);
      int  p2 = po[t];
      bool use_p = (p2 >= 0) && ((float)t <= (float)p2 + ht);
      // observed positions are fixed points -> in-place fill is safe
      xo[t] = use_n ? xo[n] : (use_p ? xo[p2] : xv);
    }
  }
}

// ---------------- stage 3: persistent WMMA GRU scan -------------------------
// LDS per WG: sH 128 KB + sW 96 KB + sGH 48 KB + sHN 8 KB = 280 KB < 320 KB/WGP
__global__ __launch_bounds__(THREADS)
void gru_kernel(const float* __restrict__ inp,  const float* __restrict__ w_ih,
                const float* __restrict__ w_hh, const float* __restrict__ b_ih,
                const float* __restrict__ b_hh, const float* __restrict__ xs,
                float* __restrict__ h0f, float* __restrict__ h1f,
                bf16* __restrict__ hb0, bf16* __restrict__ hb1, unsigned* sync) {
  extern __shared__ char smem[];
  bf16*  sH  = (bf16*)smem;                               // 128x512 bf16 = 128 KB
  bf16*  sW  = (bf16*)(smem + Bsz * Hdim * 2);            // swizzled B-frags, 96 KB
  float* sGH = (float*)(smem + Bsz * Hdim * 2 + NT * KT * 32 * 16 * 2); // 128x96 f32
  bf16*  sHN = (bf16*)(smem + Bsz * Hdim * 2 + NT * KT * 32 * 16 * 2 + Bsz * GPW * 4); // 128x32 bf16

  const int tid  = threadIdx.x;
  const int lane = tid & 31;
  const int wave = tid >> 5;
  const int wg   = blockIdx.x;
  const int cid  = __builtin_amdgcn_cluster_id_x();

  // hint: pull the weight slice toward this WGP (global_prefetch_b8)
  for (int r = tid; r < GPW; r += THREADS) {
    int gate = (r >> 5) * Hdim + wg * HPW + (r & 31);
    __builtin_prefetch(w_hh + (size_t)gate * Hdim, 0, 3);
  }

  // Stage w_hh slice into LDS, pre-swizzled into WMMA B-fragment order:
  // B 32x16 bf16: lane = half*16 + n, element j -> K = half*8 + (j&7) + (j>>3)*16  (matches A)
  for (int i = tid; i < NT * KT * 32 * 16; i += THREADS) {
    int j  = i & 15;
    int ln = (i >> 4) & 31;
    int kt = (i >> 9) & 15;
    int nt = i >> 13;
    int nl = nt * 16 + (ln & 15);                         // local gate col 0..95
    int K  = kt * 32 + ((ln >> 4) << 3) + (j & 7) + ((j >> 3) << 4);
    int gate = (nl >> 5) * Hdim + wg * HPW + (nl & 31);   // r | z | n blocks
    sW[i] = (bf16)w_hh[(size_t)gate * Hdim + K];
  }

  // zero initial hidden state: fp32 master + bf16 mirror (ws is poisoned)
  for (int i = wg * THREADS + tid; i < Bsz * Hdim; i += NWG * THREADS) h0f[i] = 0.f;
  for (int i = wg * THREADS + tid; i < (Bsz * Hdim) / 2; i += NWG * THREADS)
    ((unsigned*)hb0)[i] = 0u;
  grid_barrier(sync, sync + 1, NWG, cid);

  const int arow  = wave * 16 + (lane & 15);              // A: M row
  const int ahalf = (lane >> 4) << 3;                     // A: K sub-offset

  for (int t = 0; t < Tlen; ++t) {
    const float* hcf = (t & 1) ? h1f : h0f;               // fp32 master (read)
    float*       hnf = (t & 1) ? h0f : h1f;               // fp32 master (write)
    const bf16*  hbc = (t & 1) ? hb1 : hb0;               // bf16 mirror (read)
    bf16*        hbn = (t & 1) ? hb0 : hb1;               // bf16 mirror (write)

    // snapshot h: async byte-copy of bf16 mirror, global -> LDS (no VGPR round-trip)
#if HAS_ASYNC_LDS
    {
      const char* g = (const char*)hbc;
      char*       l = (char*)sH;
      for (int i = tid * 16; i < Bsz * Hdim * 2; i += THREADS * 16)
        async_g2l_b128(g + i, l + i);
      __builtin_amdgcn_s_wait_asynccnt(0);
    }
#else
    for (int i = tid; i < Bsz * Hdim; i += THREADS) sH[i] = (bf16)hcf[i];
#endif
    __syncthreads();

    // gh_slice[128 x 96] = h[128 x 512] x w_hh_slice^T   via bf16 WMMA
    v8f acc[NT] = {};
    for (int kt = 0; kt < KT; ++kt) {
      int kb = kt * 32 + ahalf;
      // A 16x32 bf16 fragment: lanes<16 hold K {0..7,16..23}, lanes>=16 {8..15,24..31}
      v8bf lo = *(const v8bf*)(sH + arow * Hdim + kb);
      v8bf hi = *(const v8bf*)(sH + arow * Hdim + kb + 16);
      v16bf a = __builtin_shufflevector(lo, hi, 0,1,2,3,4,5,6,7,8,9,10,11,12,13,14,15);
#pragma unroll
      for (int nt = 0; nt < NT; ++nt) {
        v16bf bm = *(const v16bf*)(sW + (((nt * KT + kt) * 32 + lane) << 4));
        acc[nt] = __builtin_amdgcn_wmma_f32_16x16x32_bf16(
            false, a, false, bm, (short)0, acc[nt], false, false);
      }
    }
    // spill C tiles: VGPR v -> M = mt*16 + (lane>=16)*8 + v, N = nt*16 + lane%16
#pragma unroll
    for (int nt = 0; nt < NT; ++nt) {
      int n     = nt * 16 + (lane & 15);
      int mbase = wave * 16 + ((lane >> 4) << 3);
#pragma unroll
      for (int v = 0; v < 8; ++v) sGH[(mbase + v) * GPW + n] = acc[nt][v];
    }
    __syncthreads();

    // elementwise gate math for this WG's 32 h-columns
    for (int e = tid; e < Bsz * HPW; e += THREADS) {
      int b = e >> 5;
      int j = e & 31;
      int c = wg * HPW + j;
      float ghr = sGH[b * GPW + j]      + b_hh[c];
      float ghz = sGH[b * GPW + 32 + j] + b_hh[Hdim + c];
      float ghn = sGH[b * GPW + 64 + j] + b_hh[2 * Hdim + c];
      float x  = xs[(size_t)b * Tlen + t];
      float mk = inp[(size_t)b * 3 * Tlen + 2 * Tlen + t];
      float gir = x * w_ih[2 * c]              + mk * w_ih[2 * c + 1]              + b_ih[c];
      float giz = x * w_ih[2 * (Hdim + c)]     + mk * w_ih[2 * (Hdim + c) + 1]     + b_ih[Hdim + c];
      float gin = x * w_ih[2 * (2 * Hdim + c)] + mk * w_ih[2 * (2 * Hdim + c) + 1] + b_ih[2 * Hdim + c];
      float r  = 1.0f / (1.0f + __expf(-(gir + ghr)));
      float z  = 1.0f / (1.0f + __expf(-(giz + ghz)));
      float ng = tanhf(gin + r * ghn);                    // b_hh_n scaled by r, as in ref
      float hp = hcf[(size_t)b * Hdim + c];               // fp32 recurrence accumulation
      float hnew = (1.0f - z) * ng + z * hp;
      hnf[(size_t)b * Hdim + c] = hnew;
#if HAS_ASYNC_LDS
      sHN[e] = (bf16)hnew;                                // stage bf16 slice for async store
#else
      hbn[(size_t)b * Hdim + c] = (bf16)hnew;
#endif
    }
#if HAS_ASYNC_LDS
    __syncthreads();
    {
      // push the bf16 slice out with GLOBAL_STORE_ASYNC_FROM_LDS_B128
      // LDS: 128 rows x 64 B contiguous; global: row stride Hdim*2 B
      for (int i = tid; i < (Bsz * HPW * 2) / 16; i += THREADS) {
        int b = i >> 2;                                   // 4 x 16B chunks per row
        int q = i & 3;
        async_l2g_b128((char*)hbn + ((size_t)b * Hdim + wg * HPW) * 2 + q * 16,
                       (const char*)sHN + (size_t)b * HPW * 2 + q * 16);
      }
      __builtin_amdgcn_s_wait_asynccnt(0);
    }
#endif
    grid_barrier(sync, sync + 1, NWG, cid);               // readers done -> writers may proceed
  }
}

// ---------------- stage 4: final FC (OUT=1) ---------------------------------
__global__ void fc_kernel(const float* __restrict__ h, const float* __restrict__ w_fc,
                          const float* __restrict__ b_fc, float* __restrict__ out) {
  int b = threadIdx.x;
  if (b >= Bsz) return;
  float s = b_fc[0];
  for (int k = 0; k < Hdim; ++k) s += h[(size_t)b * Hdim + k] * w_fc[k];
  out[b] = s;
}

extern "C" void kernel_launch(void* const* d_in, const int* in_sizes, int n_in,
                              void* d_out, int out_size, void* d_ws, size_t ws_size,
                              hipStream_t stream) {
  (void)in_sizes; (void)n_in; (void)out_size; (void)ws_size;
  const float* inp  = (const float*)d_in[0];
  const float* w_ih = (const float*)d_in[1];
  const float* w_hh = (const float*)d_in[2];
  const float* b_ih = (const float*)d_in[3];
  const float* b_hh = (const float*)d_in[4];
  const float* w_fc = (const float*)d_in[5];
  const float* b_fc = (const float*)d_in[6];
  float* out = (float*)d_out;

  // workspace carve (all segments 16B-aligned)
  float*    xs   = (float*)d_ws;                 // B*T fp32
  int*      pb   = (int*)(xs + Bsz * Tlen);      // B*T i32
  float*    h0f  = (float*)(pb + Bsz * Tlen);    // B*H fp32 master (ping)
  float*    h1f  = h0f + Bsz * Hdim;             // B*H fp32 master (pong)
  bf16*     hb0  = (bf16*)(h1f + Bsz * Hdim);    // B*H bf16 mirror (ping)
  bf16*     hb1  = hb0 + Bsz * Hdim;             // B*H bf16 mirror (pong)
  float*    scal = (float*)(hb1 + Bsz * Hdim);   // [half_tau, trigger]
  unsigned* sync = (unsigned*)(scal + 4);        // [cnt, gen]

  prep_kernel<<<1, THREADS, 0, stream>>>(w_ih, scal, sync);
  fill_kernel<<<1, 128, 0, stream>>>(inp, scal, xs, pb);

  size_t lds = (size_t)Bsz * Hdim * 2            // sH  128 KB
             + (size_t)NT * KT * 32 * 16 * 2     // sW   96 KB
             + (size_t)Bsz * GPW * 4             // sGH  48 KB
             + (size_t)Bsz * HPW * 2;            // sHN   8 KB  => 280 KB < 320 KB/WGP
  gru_kernel<<<NWG, THREADS, lds, stream>>>(inp, w_ih, w_hh, b_ih, b_hh, xs,
                                            h0f, h1f, hb0, hb1, sync);

  fc_kernel<<<1, 128, 0, stream>>>(h0f, w_fc, b_fc, out);  // T even -> final h in h0f
}
